// HModel_71055938945440
// MI455X (gfx1250) — matrix operand order repaired and checked by reading, not compiled, not run on hardware
//
#include <hip/hip_runtime.h>
#include <math.h>

// ---------------------------------------------------------------------------
// GNN forward (GINE, GINE, GATv2, TransformerConv x2, GINE) for gfx1250.
// Dense projections: LDS-tiled wave32 WMMA GEMM, 4x v_wmma_f32_16x16x32_f16
// per K-step per wave with A/B fragment reuse; packed b32 LDS staging.
// Edge aggregation / segment softmax: fp32 L2 atomics (node tables << L2).
// ---------------------------------------------------------------------------

#define NN 20000      // nodes
#define NE 320000     // edges
#define NH 4          // transformer heads

typedef __attribute__((ext_vector_type(16))) _Float16 v16h;
typedef __attribute__((ext_vector_type(8)))  float    v8f;

// ---------------- float atomic max (sign-aware int trick) ------------------
__device__ __forceinline__ void atomicMaxF(float* addr, float val) {
  if (val >= 0.0f) atomicMax((int*)addr, __float_as_int(val));
  else             atomicMin((unsigned int*)addr, __float_as_uint(val));
}

__device__ __forceinline__ float sigmoid_fast(float v) {
  return __builtin_amdgcn_rcpf(1.0f + __expf(-v));
}

__device__ __forceinline__ unsigned pack_f16x2(float a, float b) {
  union { unsigned u; _Float16 h[2]; } p;
  p.h[0] = (_Float16)a;            // fuses to v_cvt_pk_f16_f32
  p.h[1] = (_Float16)b;
  return p.u;
}

// ---------------- LDS-tiled WMMA GEMM --------------------------------------
// C[M,N] = act(A[M,K] @ W[K,N] + bias).  K multiple of 32.
// Block: 256 threads = 8 waves, tile 128(M) x 64(N); wave tile 32x32
// (2x2 WMMA 16x16x32 with fragment reuse). Row/col clamping keeps EXEC
// all-ones through WMMA; stores are guarded.
#define BM 128
#define BN 64
#define BK 32
#define LDS_STR 34            // halves; word stride 17 (coprime 64 banks)

__global__ __launch_bounds__(256)
void gemm_wmma_tiled(const float* __restrict__ A, const float* __restrict__ W,
                     const float* __restrict__ bias, float* __restrict__ C,
                     int M, int K, int N, int act) {
  __shared__ _Float16 sA[BM * LDS_STR];   // [row][k]
  __shared__ _Float16 sB[BN * LDS_STR];   // [n][k]  (transposed)

  const int tid  = threadIdx.x;
  const int lane = tid & 31;
  const int wid  = tid >> 5;
  const int wm   = wid & 3;               // wave m index 0..3
  const int wn   = wid >> 2;              // wave n index 0..1
  const int hi   = lane >> 4;
  const int l15  = lane & 15;

  const int nblk = (N + BN - 1) / BN;
  const int row0 = (blockIdx.x / nblk) * BM;
  const int col0 = (blockIdx.x % nblk) * BN;

  // staging work assignment
  const int ar  = tid >> 1;               // A: row 0..127
  const int ak  = (tid & 1) * 16;         // A: 16 contiguous k
  const int bn  = tid & 63;               // B: one column n
  const int bk0 = (tid >> 6) * 8;         // B: 8 consecutive k

  v8f acc00 = {}, acc01 = {}, acc10 = {}, acc11 = {};

  for (int k0 = 0; k0 < K; k0 += BK) {
    __syncthreads();
    {   // stage A tile (coalesced f32 loads -> packed f16x2 b32 LDS stores)
      int grow = row0 + ar; if (grow > M - 1) grow = M - 1;
      const float* ap = A + (size_t)grow * K + k0 + ak;
      if (k0 + BK < K) __builtin_prefetch(ap + BK, 0, 1);
      unsigned* dst = (unsigned*)&sA[ar * LDS_STR + ak];
#pragma unroll
      for (int i = 0; i < 8; ++i)
        dst[i] = pack_f16x2(ap[2 * i], ap[2 * i + 1]);
    }
    {   // stage B tile transposed: thread owns column n, 8 consecutive k
      int gn = col0 + bn; if (gn > N - 1) gn = N - 1;
      const float* bp = W + (size_t)(k0 + bk0) * N + gn;
      unsigned* dst = (unsigned*)&sB[bn * LDS_STR + bk0];
#pragma unroll
      for (int i = 0; i < 4; ++i)
        dst[i] = pack_f16x2(bp[(size_t)(2 * i) * N], bp[(size_t)(2 * i + 1) * N]);
    }
    __syncthreads();

    union { v16h v; unsigned u[8]; } a0, a1, b0, b1;
    const int r0 = wm * 32 + l15;
    const int n0 = wn * 32 + l15;
#pragma unroll
    for (int j = 0; j < 8; ++j) {
      // A 16x32 f16 fragment: lanes0-15 K0..15, lanes16-31 K16..31.
      // Pairs contiguous -> one b32 LDS read.
      int ka = 2 * j + (hi ? 16 : 0);
      a0.u[j] = *(const unsigned*)&sA[r0 * LDS_STR + ka];
      a1.u[j] = *(const unsigned*)&sA[(r0 + 16) * LDS_STR + ka];
      // B 32x16 f16 fragment: lanes0-15 K0..15, lanes16-31 K16..31.
      int kb = 2 * j + (hi ? 16 : 0);
      b0.u[j] = *(const unsigned*)&sB[n0 * LDS_STR + kb];
      b1.u[j] = *(const unsigned*)&sB[(n0 + 16) * LDS_STR + kb];
    }
    acc00 = __builtin_amdgcn_wmma_f32_16x16x32_f16(false, a0.v, false, b0.v,
                                                   (short)0, acc00, false, false);
    acc01 = __builtin_amdgcn_wmma_f32_16x16x32_f16(false, a0.v, false, b1.v,
                                                   (short)0, acc01, false, false);
    acc10 = __builtin_amdgcn_wmma_f32_16x16x32_f16(false, a1.v, false, b0.v,
                                                   (short)0, acc10, false, false);
    acc11 = __builtin_amdgcn_wmma_f32_16x16x32_f16(false, a1.v, false, b1.v,
                                                   (short)0, acc11, false, false);
  }

  auto st = [&](const v8f& acc, int mbase, int nbase) {
    int n = nbase + l15;
    if (n >= N) return;
#pragma unroll
    for (int i = 0; i < 8; ++i) {
      int m = mbase + i + (hi ? 8 : 0);
      if (m < M) {
        float v = acc[i];
        if (bias) v += bias[n];
        if (act == 1) v = sigmoid_fast(v);
        C[(size_t)m * N + n] = v;
      }
    }
  };
  st(acc00, row0 + wm * 32,      col0 + wn * 32);
  st(acc01, row0 + wm * 32,      col0 + wn * 32 + 16);
  st(acc10, row0 + wm * 32 + 16, col0 + wn * 32);
  st(acc11, row0 + wm * 32 + 16, col0 + wn * 32 + 16);
}

// ---------------- elementwise helpers --------------------------------------
__global__ void fill_kernel(float* y, float v, long n) {
  long i = blockIdx.x * (long)blockDim.x + threadIdx.x;
  if (i < n) y[i] = v;
}

__global__ void scale_init_kernel(const float* __restrict__ x,
                                  const float* __restrict__ eps,
                                  float* __restrict__ y, long n) {
  long i = blockIdx.x * (long)blockDim.x + threadIdx.x;
  if (i < n) y[i] = (1.0f + eps[0]) * x[i];
}

__global__ void bias_bcast_kernel(const float* __restrict__ b,
                                  float* __restrict__ y, int C, long n) {
  long i = blockIdx.x * (long)blockDim.x + threadIdx.x;
  if (i < n) y[i] = b[i % C];
}

// ---------------- GINE: agg[tgt] += relu(x[src] + e) -----------------------
__global__ void gine_edge_kernel(const float* __restrict__ x,
                                 const float* __restrict__ e,
                                 const int* __restrict__ src,
                                 const int* __restrict__ tgt,
                                 float* __restrict__ agg, int C, long n) {
  long idx = blockIdx.x * (long)blockDim.x + threadIdx.x;
  if (idx >= n) return;
  int ed = (int)(idx / C), c = (int)(idx % C);
  float m = x[(size_t)src[ed] * C + c] + e[idx];
  if (m > 0.0f) atomicAdd(&agg[(size_t)tgt[ed] * C + c], m);
}

// ---------------- GATv2 -----------------------------------------------------
__global__ void gat_cnt_ea_kernel(const float* __restrict__ ea,
                                  const int* __restrict__ tgt,
                                  float* __restrict__ cnt,
                                  float* __restrict__ sum_ea, long n) {
  long idx = blockIdx.x * (long)blockDim.x + threadIdx.x;
  if (idx >= n) return;
  int e = (int)(idx >> 5), c = (int)(idx & 31);
  atomicAdd(&sum_ea[(size_t)tgt[e] * 32 + c], ea[idx]);
  if (c == 0) atomicAdd(&cnt[tgt[e]], 1.0f);
}

__global__ void gat_mean_kernel(float* __restrict__ mea,
                                const float* __restrict__ cnt, long n) {
  long idx = blockIdx.x * (long)blockDim.x + threadIdx.x;
  if (idx >= n) return;
  mea[idx] = mea[idx] / fmaxf(cnt[idx >> 5], 1.0f);
}

__global__ void gat_logit_edge_kernel(const float* __restrict__ xl,
                                      const float* __restrict__ xr,
                                      const float* __restrict__ eE,
                                      const float* __restrict__ att,
                                      const int* __restrict__ src,
                                      const int* __restrict__ tgt,
                                      float* __restrict__ logit,
                                      float* __restrict__ segm, int E) {
  int e = blockIdx.x * blockDim.x + threadIdx.x;
  if (e >= E) return;
  const float* a = xl + (size_t)src[e] * 128;
  const float* b = xr + (size_t)tgt[e] * 128;
  const float* c = eE + (size_t)e * 128;
  float s = 0.0f;
  for (int i = 0; i < 128; ++i) {
    float z = a[i] + b[i] + c[i];
    z = (z > 0.0f) ? z : 0.2f * z;
    s += z * att[i];
  }
  logit[e] = s;
  atomicMaxF(&segm[tgt[e]], s);
}

__global__ void gat_logit_self_kernel(const float* __restrict__ xl,
                                      const float* __restrict__ xr,
                                      const float* __restrict__ eN,
                                      const float* __restrict__ att,
                                      float* __restrict__ selfv,
                                      float* __restrict__ segm, int N) {
  int nd = blockIdx.x * blockDim.x + threadIdx.x;
  if (nd >= N) return;
  const float* a = xl + (size_t)nd * 128;
  const float* b = xr + (size_t)nd * 128;
  const float* c = eN + (size_t)nd * 128;
  float s = 0.0f;
  for (int i = 0; i < 128; ++i) {
    float z = a[i] + b[i] + c[i];
    z = (z > 0.0f) ? z : 0.2f * z;
    s += z * att[i];
  }
  selfv[nd] = s;
  atomicMaxF(&segm[nd], s);
}

__global__ void gat_exp_edge_kernel(float* __restrict__ logit,
                                    const float* __restrict__ segm,
                                    float* __restrict__ segs,
                                    const int* __restrict__ tgt, int E) {
  int e = blockIdx.x * blockDim.x + threadIdx.x;
  if (e >= E) return;
  float ex = __expf(logit[e] - segm[tgt[e]]);
  logit[e] = ex;
  atomicAdd(&segs[tgt[e]], ex);
}

__global__ void gat_exp_self_kernel(float* __restrict__ selfv,
                                    const float* __restrict__ segm,
                                    float* __restrict__ segs, int N) {
  int nd = blockIdx.x * blockDim.x + threadIdx.x;
  if (nd >= N) return;
  float ex = __expf(selfv[nd] - segm[nd]);
  selfv[nd] = ex;
  atomicAdd(&segs[nd], ex);
}

__global__ void gat_agg_edge_kernel(const float* __restrict__ logit,
                                    const float* __restrict__ segs,
                                    const float* __restrict__ xl,
                                    const int* __restrict__ src,
                                    const int* __restrict__ tgt,
                                    float* __restrict__ out, long n) {
  long idx = blockIdx.x * (long)blockDim.x + threadIdx.x;
  if (idx >= n) return;
  int e = (int)(idx >> 7), c = (int)(idx & 127);
  float a = logit[e] / (segs[tgt[e]] + 1e-16f);
  atomicAdd(&out[(size_t)tgt[e] * 128 + c], a * xl[(size_t)src[e] * 128 + c]);
}

__global__ void gat_agg_self_kernel(const float* __restrict__ selfv,
                                    const float* __restrict__ segs,
                                    const float* __restrict__ xl,
                                    float* __restrict__ out, long n) {
  long idx = blockIdx.x * (long)blockDim.x + threadIdx.x;
  if (idx >= n) return;
  int nd = (int)(idx >> 7), c = (int)(idx & 127);
  float a = selfv[nd] / (segs[nd] + 1e-16f);
  atomicAdd(&out[(size_t)nd * 128 + c], a * xl[(size_t)nd * 128 + c]);
}

// ---------------- TransformerConv (H=4, c=64, HC=256) ----------------------
__global__ void tr_alpha_kernel(const float* __restrict__ q,
                                const float* __restrict__ k,
                                const float* __restrict__ eb,
                                const int* __restrict__ src,
                                const int* __restrict__ tgt,
                                float* __restrict__ logit,
                                float* __restrict__ segm, long n) {
  long idx = blockIdx.x * (long)blockDim.x + threadIdx.x;
  if (idx >= n) return;
  int e = (int)(idx >> 2), h = (int)(idx & 3);
  const float* qp = q + (size_t)tgt[e] * 256 + h * 64;
  const float* kp = k + (size_t)src[e] * 256 + h * 64;
  const float* ep = eb + (size_t)e * 256 + h * 64;
  float s = 0.0f;
  for (int c = 0; c < 64; ++c) s += qp[c] * (kp[c] + ep[c]);
  s *= 0.125f;                       // 1/sqrt(64)
  logit[idx] = s;
  atomicMaxF(&segm[(size_t)tgt[e] * NH + h], s);
}

__global__ void tr_exp_kernel(float* __restrict__ logit,
                              const float* __restrict__ segm,
                              float* __restrict__ segs,
                              const int* __restrict__ tgt, long n) {
  long idx = blockIdx.x * (long)blockDim.x + threadIdx.x;
  if (idx >= n) return;
  int e = (int)(idx >> 2), h = (int)(idx & 3);
  float ex = __expf(logit[idx] - segm[(size_t)tgt[e] * NH + h]);
  logit[idx] = ex;
  atomicAdd(&segs[(size_t)tgt[e] * NH + h], ex);
}

__global__ void tr_agg_kernel(const float* __restrict__ logit,
                              const float* __restrict__ segs,
                              const float* __restrict__ v,
                              const float* __restrict__ eb,
                              const int* __restrict__ src,
                              const int* __restrict__ tgt,
                              float* __restrict__ out, long n) {
  long idx = blockIdx.x * (long)blockDim.x + threadIdx.x;
  if (idx >= n) return;
  int e = (int)(idx >> 8), c = (int)(idx & 255), h = c >> 6;
  float a = logit[(size_t)e * NH + h] / (segs[(size_t)tgt[e] * NH + h] + 1e-16f);
  atomicAdd(&out[(size_t)tgt[e] * 256 + c],
            a * (v[(size_t)src[e] * 256 + c] + eb[idx]));
}

// ---------------------------------------------------------------------------
static inline void launch_gemm(const float* A, const float* W, const float* b,
                               float* C, int M, int K, int N, int act,
                               hipStream_t s) {
  int blocks = ((M + BM - 1) / BM) * ((N + BN - 1) / BN);
  gemm_wmma_tiled<<<blocks, 256, 0, s>>>(A, W, b, C, M, K, N, act);
}

#define GRID1D(n) ((int)(((n) + 255) / 256)), 256, 0, stream

extern "C" void kernel_launch(void* const* d_in, const int* in_sizes, int n_in,
                              void* d_out, int out_size, void* d_ws,
                              size_t ws_size, hipStream_t stream) {
  (void)in_sizes; (void)n_in; (void)out_size; (void)ws_size;

  // ---- inputs (recursive dict-insertion order) ----
  const float* x  = (const float*)d_in[0];     // [N,64]
  const float* ea = (const float*)d_in[1];     // [E,32]
  const float *g1_eW = (const float*)d_in[2],  *g1_eb = (const float*)d_in[3];
  const float *g1_eps = (const float*)d_in[4];
  const float *g1_W1 = (const float*)d_in[5],  *g1_b1 = (const float*)d_in[6];
  const float *g1_W2 = (const float*)d_in[7],  *g1_b2 = (const float*)d_in[8];
  const float *g2_eW = (const float*)d_in[9],  *g2_eb = (const float*)d_in[10];
  const float *g2_eps = (const float*)d_in[11];
  const float *g2_W1 = (const float*)d_in[12], *g2_b1 = (const float*)d_in[13];
  const float *g2_W2 = (const float*)d_in[14], *g2_b2 = (const float*)d_in[15];
  const float *gat_lW = (const float*)d_in[16], *gat_lb = (const float*)d_in[17];
  const float *gat_rW = (const float*)d_in[18], *gat_rb = (const float*)d_in[19];
  const float *gat_eW = (const float*)d_in[20], *gat_att = (const float*)d_in[21];
  const float *gat_bias = (const float*)d_in[22];
  const float *t4_qW = (const float*)d_in[23], *t4_qb = (const float*)d_in[24];
  const float *t4_kW = (const float*)d_in[25], *t4_kb = (const float*)d_in[26];
  const float *t4_vW = (const float*)d_in[27], *t4_vb = (const float*)d_in[28];
  const float *t4_eW = (const float*)d_in[29], *t4_eb = (const float*)d_in[30];
  const float *t4_sW = (const float*)d_in[31], *t4_sb = (const float*)d_in[32];
  const float *t5_qW = (const float*)d_in[33], *t5_qb = (const float*)d_in[34];
  const float *t5_kW = (const float*)d_in[35], *t5_kb = (const float*)d_in[36];
  const float *t5_vW = (const float*)d_in[37], *t5_vb = (const float*)d_in[38];
  const float *t5_eW = (const float*)d_in[39], *t5_eb = (const float*)d_in[40];
  const float *t5_sW = (const float*)d_in[41], *t5_sb = (const float*)d_in[42];
  const float *g6_eW = (const float*)d_in[43], *g6_eb = (const float*)d_in[44];
  const float *g6_eps = (const float*)d_in[45];
  const float *g6_W1 = (const float*)d_in[46], *g6_b1 = (const float*)d_in[47];
  const float *g6_W2 = (const float*)d_in[48], *g6_b2 = (const float*)d_in[49];
  const int* ei  = (const int*)d_in[50];       // [2,E]
  const int* src = ei;
  const int* tgt = ei + NE;

  float* out = (float*)d_out;                  // [N,64]

  // ---- workspace layout ----
  float* f = (float*)d_ws;
  size_t o = 0;
  const size_t NB = (size_t)NN * 256;
  float* nb1  = f + o; o += NB;
  float* nb2  = f + o; o += NB;
  float* nb3  = f + o; o += NB;
  float* nb4  = f + o; o += NB;
  float* nb5  = f + o; o += NB;
  float* nb6  = f + o; o += NB;
  float* ebuf = f + o; o += (size_t)NE * 256;
  float* elog = f + o; o += (size_t)NE * NH;
  float* segm = f + o; o += (size_t)NN * NH;
  float* segs = f + o; o += (size_t)NN * NH;
  float* cnt  = f + o; o += (size_t)NN;
  float* mea  = f + o; o += (size_t)NN * 32;
  float* eN   = f + o; o += (size_t)NN * 128;
  float* selfv= f + o; o += (size_t)NN;

  // ================= g1: GINE (64 -> 64) =================
  launch_gemm(ea, g1_eW, g1_eb, ebuf, NE, 32, 64, 0, stream);
  scale_init_kernel<<<GRID1D((long)NN * 64)>>>(x, g1_eps, nb1, (long)NN * 64);
  gine_edge_kernel<<<GRID1D((long)NE * 64)>>>(x, ebuf, src, tgt, nb1, 64,
                                              (long)NE * 64);
  launch_gemm(nb1, g1_W1, g1_b1, nb2, NN, 64, 64, 1, stream);
  launch_gemm(nb2, g1_W2, g1_b2, nb3, NN, 64, 64, 0, stream);   // x1 = nb3

  // ================= g2: GINE (64 -> 64) =================
  launch_gemm(ea, g2_eW, g2_eb, ebuf, NE, 32, 64, 0, stream);
  scale_init_kernel<<<GRID1D((long)NN * 64)>>>(nb3, g2_eps, nb1, (long)NN * 64);
  gine_edge_kernel<<<GRID1D((long)NE * 64)>>>(nb3, ebuf, src, tgt, nb1, 64,
                                              (long)NE * 64);
  launch_gemm(nb1, g2_W1, g2_b1, nb2, NN, 64, 32, 1, stream);
  launch_gemm(nb2, g2_W2, g2_b2, nb4, NN, 32, 64, 0, stream);   // x2 = nb4

  // ================= gat: GATv2 (64 -> 128) =================
  fill_kernel<<<GRID1D((long)NN)>>>(cnt, 0.0f, (long)NN);
  fill_kernel<<<GRID1D((long)NN * 32)>>>(mea, 0.0f, (long)NN * 32);
  gat_cnt_ea_kernel<<<GRID1D((long)NE * 32)>>>(ea, tgt, cnt, mea, (long)NE * 32);
  gat_mean_kernel<<<GRID1D((long)NN * 32)>>>(mea, cnt, (long)NN * 32);
  launch_gemm(nb4, gat_lW, gat_lb, nb1, NN, 64, 128, 0, stream);   // xl
  launch_gemm(nb4, gat_rW, gat_rb, nb2, NN, 64, 128, 0, stream);   // xr
  launch_gemm(ea,  gat_eW, nullptr, ebuf, NE, 32, 128, 0, stream); // e (edges)
  launch_gemm(mea, gat_eW, nullptr, eN,   NN, 32, 128, 0, stream); // e (self)
  fill_kernel<<<GRID1D((long)NN)>>>(segm, -INFINITY, (long)NN);
  fill_kernel<<<GRID1D((long)NN)>>>(segs, 0.0f, (long)NN);
  gat_logit_edge_kernel<<<GRID1D((long)NE)>>>(nb1, nb2, ebuf, gat_att, src, tgt,
                                              elog, segm, NE);
  gat_logit_self_kernel<<<GRID1D((long)NN)>>>(nb1, nb2, eN, gat_att, selfv,
                                              segm, NN);
  gat_exp_edge_kernel<<<GRID1D((long)NE)>>>(elog, segm, segs, tgt, NE);
  gat_exp_self_kernel<<<GRID1D((long)NN)>>>(selfv, segm, segs, NN);
  bias_bcast_kernel<<<GRID1D((long)NN * 128)>>>(gat_bias, nb3, 128,
                                                (long)NN * 128);
  gat_agg_edge_kernel<<<GRID1D((long)NE * 128)>>>(elog, segs, nb1, src, tgt,
                                                  nb3, (long)NE * 128);
  gat_agg_self_kernel<<<GRID1D((long)NN * 128)>>>(selfv, segs, nb1, nb3,
                                                  (long)NN * 128);
  // x3 = nb3 [N,128]

  // ================= t4: TransformerConv (128 -> 256) =================
  launch_gemm(nb3, t4_qW, t4_qb, nb1, NN, 128, 256, 0, stream);  // q
  launch_gemm(nb3, t4_kW, t4_kb, nb2, NN, 128, 256, 0, stream);  // k
  launch_gemm(nb3, t4_vW, t4_vb, nb4, NN, 128, 256, 0, stream);  // v
  launch_gemm(nb3, t4_sW, t4_sb, nb5, NN, 128, 256, 0, stream);  // skip -> out
  launch_gemm(ea,  t4_eW, t4_eb, ebuf, NE, 32, 256, 0, stream);  // e
  fill_kernel<<<GRID1D((long)NN * NH)>>>(segm, -INFINITY, (long)NN * NH);
  fill_kernel<<<GRID1D((long)NN * NH)>>>(segs, 0.0f, (long)NN * NH);
  tr_alpha_kernel<<<GRID1D((long)NE * NH)>>>(nb1, nb2, ebuf, src, tgt, elog,
                                             segm, (long)NE * NH);
  tr_exp_kernel<<<GRID1D((long)NE * NH)>>>(elog, segm, segs, tgt, (long)NE * NH);
  tr_agg_kernel<<<GRID1D((long)NE * 256)>>>(elog, segs, nb4, ebuf, src, tgt,
                                            nb5, (long)NE * 256);
  // x4 = nb5 [N,256]

  // ================= t5: TransformerConv (256 -> 256) =================
  launch_gemm(nb5, t5_qW, t5_qb, nb1, NN, 256, 256, 0, stream);  // q
  launch_gemm(nb5, t5_kW, t5_kb, nb2, NN, 256, 256, 0, stream);  // k
  launch_gemm(nb5, t5_vW, t5_vb, nb4, NN, 256, 256, 0, stream);  // v
  launch_gemm(nb5, t5_sW, t5_sb, nb6, NN, 256, 256, 0, stream);  // skip -> out
  launch_gemm(ea,  t5_eW, t5_eb, ebuf, NE, 32, 256, 0, stream);  // e
  fill_kernel<<<GRID1D((long)NN * NH)>>>(segm, -INFINITY, (long)NN * NH);
  fill_kernel<<<GRID1D((long)NN * NH)>>>(segs, 0.0f, (long)NN * NH);
  tr_alpha_kernel<<<GRID1D((long)NE * NH)>>>(nb1, nb2, ebuf, src, tgt, elog,
                                             segm, (long)NE * NH);
  tr_exp_kernel<<<GRID1D((long)NE * NH)>>>(elog, segm, segs, tgt, (long)NE * NH);
  tr_agg_kernel<<<GRID1D((long)NE * 256)>>>(elog, segs, nb4, ebuf, src, tgt,
                                            nb6, (long)NE * 256);
  // x5 = nb6 [N,256]

  // ================= g6: GINE (256 -> 64) =================
  launch_gemm(ea, g6_eW, g6_eb, ebuf, NE, 32, 256, 0, stream);
  scale_init_kernel<<<GRID1D((long)NN * 256)>>>(nb6, g6_eps, nb1,
                                                (long)NN * 256);
  gine_edge_kernel<<<GRID1D((long)NE * 256)>>>(nb6, ebuf, src, tgt, nb1, 256,
                                               (long)NE * 256);
  launch_gemm(nb1, g6_W1, g6_b1, nb2, NN, 256, 128, 1, stream);
  launch_gemm(nb2, g6_W2, g6_b2, out, NN, 128, 64, 0, stream);
}
